// Attention_1752346657012
// MI455X (gfx1250) — matrix operand order repaired and checked by reading, not compiled, not run on hardware
//
#include <hip/hip_runtime.h>
#include <hip/hip_bf16.h>

// ---------------- problem constants ----------------
#define T_      2048
#define D_      2048
#define N_      16      // query heads
#define KH_     8       // kv heads
#define H_      128     // head dim
#define WINDOW_ 1024
#define SOFT_CAP_ 50.0f
#define INV_SOFT_CAP_ 0.02f
#define Q_SCALE_ 0.08838834764831845f     // 1/sqrt(128)
#define LOG_BASE_ 9.210340371976184f      // ln(10000)
#define K_MASK_ (-2.3819763e38f)

typedef __attribute__((ext_vector_type(16))) __bf16 v16bf;
typedef __attribute__((ext_vector_type(8)))  __bf16 v8bf;
typedef __attribute__((ext_vector_type(8)))  float  v8f;
typedef __attribute__((ext_vector_type(4)))  unsigned int v4u;
typedef __attribute__((ext_vector_type(4)))  int v4i;
typedef __attribute__((ext_vector_type(8)))  int v8i;

static __device__ inline __bf16 f2bf(float f) { return (__bf16)f; }

// ---------------- WMMA: D = A(16x32 bf16) * B(32x16 bf16) + C(16x16 f32) ----
static __device__ inline v8f wmma_bf16(v16bf a, v16bf b, v8f c) {
#if defined(__HIP_DEVICE_COMPILE__)
  return __builtin_amdgcn_wmma_f32_16x16x32_bf16(
      false, a, false, b, (short)0, c, false, false);
#else
  (void)a; (void)b;
  return c;
#endif
}

#if defined(__HIP_DEVICE_COMPILE__) && __has_builtin(__builtin_amdgcn_s_wait_tensorcnt)
#define WAIT_TENSORCNT(n) __builtin_amdgcn_s_wait_tensorcnt(n)
#else
#define WAIT_TENSORCNT(n) ((void)0)
#endif

#if defined(__HIP_DEVICE_COMPILE__) && __has_builtin(__builtin_amdgcn_tensor_load_to_lds)
#define HAVE_TDM 1
#else
#define HAVE_TDM 0
#endif

// Async TDM load of a 2-D tile (tile_d1 rows x tile_d0 elements, dim0
// contiguous in memory, row stride = stride0 elements) into LDS.
// data_size_code: 1 = 2 bytes (bf16), 2 = 4 bytes (f32).
// Elements beyond tensor_d0/tensor_d1 are zero-filled by TDM OOB handling.
static __device__ inline void tdm_load_2d(void* lds_ptr, const void* gptr,
                                          unsigned data_size_code,
                                          unsigned tile_d0, unsigned tile_d1,
                                          unsigned tensor_d0, unsigned tensor_d1,
                                          unsigned stride0, int lane) {
#if HAVE_TDM
  (void)lane;
  const unsigned lds_off = (unsigned)(size_t)lds_ptr;   // LDS aperture: low 32 bits
  const unsigned long long ga = (unsigned long long)(size_t)gptr;
  v4u g0;
  g0[0] = 1u;                                            // count=1, user mode
  g0[1] = lds_off;                                       // lds_addr
  g0[2] = (unsigned)(ga & 0xffffffffu);                  // global_addr[31:0]
  g0[3] = (unsigned)((ga >> 32) & 0x1ffffffu) | (2u << 30); // addr[56:32] | type=2
  v8i g1;
  g1[0] = (int)(data_size_code << 16);                   // mask=0, data_size
  g1[1] = (int)((tensor_d0 & 0xffffu) << 16);            // tensor_dim0[15:0]
  g1[2] = (int)((tensor_d0 >> 16) | ((tensor_d1 & 0xffffu) << 16));
  g1[3] = (int)((tensor_d1 >> 16) | (tile_d0 << 16));    // tile_dim0
  g1[4] = (int)(tile_d1 & 0xffffu);                      // tile_dim1, tile_dim2=0
  g1[5] = (int)stride0;                                  // tensor_dim0_stride[31:0]
  g1[6] = 0;
  g1[7] = 0;
  v4i z4 = {0, 0, 0, 0};
#if __has_include(<hip/amd_detail/amd_gfx1250_TDM.h>)
  v8i z8 = {0, 0, 0, 0, 0, 0, 0, 0};
  __builtin_amdgcn_tensor_load_to_lds(g0, g1, z4, z4, z8, 0);
#else
  __builtin_amdgcn_tensor_load_to_lds(g0, g1, z4, z4, 0);
#endif
#else
  if (data_size_code == 1) {
    unsigned short* d = (unsigned short*)lds_ptr;
    const unsigned short* g = (const unsigned short*)gptr;
    for (unsigned i = lane; i < tile_d0 * tile_d1; i += 32) {
      unsigned r = i / tile_d0, c = i % tile_d0;
      d[i] = (r < tensor_d1 && c < tensor_d0) ? g[(size_t)r * stride0 + c] : 0;
    }
  } else {
    unsigned* d = (unsigned*)lds_ptr;
    const unsigned* g = (const unsigned*)gptr;
    for (unsigned i = lane; i < tile_d0 * tile_d1; i += 32) {
      unsigned r = i / tile_d0, c = i % tile_d0;
      d[i] = (r < tensor_d1 && c < tensor_d0) ? g[(size_t)r * stride0 + c] : 0;
    }
  }
#endif
}

// A-matrix 16x32 bf16 fragment, vectorized: per lane two contiguous 16B runs.
static __device__ inline v16bf load_a_vec(const __bf16* __restrict__ p, int lane_kb) {
  v8bf lo = *(const v8bf*)(p + lane_kb);
  v8bf hi = *(const v8bf*)(p + lane_kb + 8);
  v16bf a;
#pragma unroll
  for (int e = 0; e < 8; ++e) { a[e] = lo[e]; a[e + 8] = hi[e]; }
  return a;
}

// ---------------- kernel 0a: f32 -> bf16 convert (same layout) ----------------
__global__ __launch_bounds__(256) void cvt_bf16_kernel(const float* __restrict__ x,
                                                       __bf16* __restrict__ xb,
                                                       int nelem) {
  int i = blockIdx.x * 256 + threadIdx.x;
  if (i < nelem) xb[i] = f2bf(x[i]);
}

// ---------------- kernel 0b: batched transpose + convert ----------------
// src: batch x [R][C] f32  ->  dst: batch x [C][R] bf16
__global__ __launch_bounds__(256) void cvtT_bf16_kernel(const float* __restrict__ src,
                                                        __bf16* __restrict__ dst,
                                                        int R, int C, int batch) {
  size_t i = (size_t)blockIdx.x * 256 + threadIdx.x;
  size_t rc = (size_t)R * C;
  if (i >= rc * (size_t)batch) return;
  int b = (int)(i / rc);
  size_t rem = i - (size_t)b * rc;
  int r = (int)(rem / C), c = (int)(rem % C);
  dst[(size_t)b * rc + (size_t)c * R + r] = f2bf(src[i]);
}

// ---------------- kernel 1: fused QKV projection + RoPE ----------------
// Weights are pre-transposed bf16: Wt[head][h][d] (row stride D_).
// grid.x = T/16 row tiles, grid.y = N + 2*KH (q | k | v heads)
__global__ __launch_bounds__(32) void proj_kernel(
    const __bf16* __restrict__ xb, const __bf16* __restrict__ wqt,
    const __bf16* __restrict__ wkt, const __bf16* __restrict__ wvt,
    __bf16* __restrict__ qb, __bf16* __restrict__ kb, __bf16* __restrict__ vt) {
  __shared__ __attribute__((aligned(32))) __bf16 wl[2][128 * 32];  // 16 KiB

  const int lane = threadIdx.x;
  const int t0   = blockIdx.x * 16;
  const int y    = blockIdx.y;

  const __bf16* Wt;
  int kind, head;
  if (y < N_)            { kind = 0; head = y;            Wt = wqt + (size_t)head * H_ * D_; }
  else if (y < N_ + KH_) { kind = 1; head = y - N_;       Wt = wkt + (size_t)head * H_ * D_; }
  else                   { kind = 2; head = y - N_ - KH_; Wt = wvt + (size_t)head * H_ * D_; }

  const int col    = lane & 15;
  const int khalf  = (lane >> 4) << 4;
  const int akb    = (lane >> 4) << 4;
  const int rowofs = (lane >> 4) << 3;
  const __bf16* arow = xb + (size_t)(t0 + (lane & 15)) * D_;

  // Wt tile: 128 h-rows x 32 d-cols (dim0 contiguous along d)
  tdm_load_2d(&wl[0][0], Wt, 1, 32, 128, D_, 128, D_, lane);

  v8f acc[8] = {};
  for (int kc = 0; kc < D_; kc += 32) {
    const int cur = (kc >> 5) & 1;
    if (kc + 32 < D_) {
      tdm_load_2d(&wl[cur ^ 1][0], Wt + kc + 32, 1, 32, 128,
                  (unsigned)(D_ - (kc + 32)), 128, D_, lane);
      WAIT_TENSORCNT(1);
    } else {
      WAIT_TENSORCNT(0);
    }
    v16bf a = load_a_vec(arow + kc, akb);
    const __bf16* wt = &wl[cur][0];
#pragma unroll
    for (int j = 0; j < 8; ++j) {
      v16bf b = *(const v16bf*)&wt[(size_t)(j * 16 + col) * 32 + khalf];
      acc[j] = wmma_bf16(a, b, acc[j]);
    }
  }

  if (kind == 2) {  // V: no RoPE; store TRANSPOSED: vt[(head*H + h)*T + t]
#pragma unroll
    for (int r = 0; r < 8; ++r) {
      int t = t0 + r + rowofs;
#pragma unroll
      for (int j = 0; j < 8; ++j)
        vt[((size_t)head * H_ + j * 16 + col) * T_ + t] = f2bf(acc[j][r]);
    }
    return;
  }

  // RoPE in C-layout: element h = 16*j+col pairs with h+64 (accumulator j+4)
  const float scale = (kind == 0) ? Q_SCALE_ : 1.0f;
#pragma unroll
  for (int sub = 0; sub < 4; ++sub) {
    const int h1 = 16 * sub + col;  // 0..63
    const float inv_ts = __expf(-(2.0f * (float)h1 / (float)H_) * LOG_BASE_);
#pragma unroll
    for (int r = 0; r < 8; ++r) {
      const int t = t0 + r + rowofs;
      float sn, cs;
      __sincosf((float)t * inv_ts, &sn, &cs);
      const float f  = acc[sub][r];
      const float s2 = acc[sub + 4][r];
      acc[sub][r]     = (f * cs - s2 * sn) * scale;
      acc[sub + 4][r] = (s2 * cs + f * sn) * scale;
    }
  }
  __bf16* dst = (kind == 0) ? qb : kb;
  const int hs = (kind == 0) ? N_ : KH_;
#pragma unroll
  for (int r = 0; r < 8; ++r) {
    int t = t0 + r + rowofs;
#pragma unroll
    for (int j = 0; j < 8; ++j)
      dst[((size_t)t * hs + head) * H_ + j * 16 + col] = f2bf(acc[j][r]);
  }
}

// ---------------- kernel 2: flash attention (16 q-rows x 32 keys / step) ----
// K from kb [t][kh][H]; V from transposed vt [kh][h][t].
__global__ __launch_bounds__(32) void attn_kernel(
    const __bf16* __restrict__ qb, const __bf16* __restrict__ kb,
    const __bf16* __restrict__ vt, __bf16* __restrict__ eb) {
  __shared__ __attribute__((aligned(32))) __bf16 ktile[2][32 * 128];  // 16 KiB
  __shared__ __attribute__((aligned(32))) __bf16 vtile[2][128 * 32];  // 16 KiB
  __shared__ __attribute__((aligned(32))) __bf16 pls[16 * 32];

  const int lane   = threadIdx.x;
  const int t0     = blockIdx.x * 16;
  const int n      = blockIdx.y;
  const int khead  = n >> 1;            // G = N/KH = 2
  const int col    = lane & 15;
  const int half   = lane >> 4;
  const int rowofs = half << 3;
  const int akb    = half << 4;

  // Q fragments (A-layout), H=128 -> 4 chunks of K=32
  v16bf qa[4];
  {
    const __bf16* qp = qb + ((size_t)(t0 + (lane & 15)) * N_ + n) * H_;
#pragma unroll
    for (int c = 0; c < 4; ++c) qa[c] = load_a_vec(qp + c * 32, akb);
  }

  v8f oacc[8] = {};
  float m[8], l[8];
#pragma unroll
  for (int r = 0; r < 8; ++r) { m[r] = -INFINITY; l[r] = 0.0f; }

  int s_lo = t0 - (WINDOW_ - 1);
  if (s_lo < 0) s_lo = 0;
  s_lo &= ~31;
  const int s_hi = t0 + 15;

  const __bf16* vrow = vt + (size_t)khead * H_ * T_;
  // K tile: 32 key-rows x 128 h (dim0 along h); V tile: 128 h-rows x 32 keys
  tdm_load_2d(&ktile[0][0], kb + ((size_t)s_lo * KH_ + khead) * H_, 1, 128, 32,
              128, (unsigned)(T_ - s_lo), KH_ * H_, lane);
  tdm_load_2d(&vtile[0][0], vrow + s_lo, 1, 32, 128,
              (unsigned)(T_ - s_lo), 128, T_, lane);

  int it = 0;
  for (int s0 = s_lo; s0 <= s_hi; s0 += 32, ++it) {
    const int cur = it & 1;
    if (s0 + 32 <= s_hi) {
      const int sn0 = s0 + 32;
      tdm_load_2d(&ktile[cur ^ 1][0], kb + ((size_t)sn0 * KH_ + khead) * H_, 1,
                  128, 32, 128, (unsigned)(T_ - sn0), KH_ * H_, lane);
      tdm_load_2d(&vtile[cur ^ 1][0], vrow + sn0, 1, 32, 128,
                  (unsigned)(T_ - sn0), 128, T_, lane);
      WAIT_TENSORCNT(2);
    } else {
      WAIT_TENSORCNT(0);
    }
    const __bf16* kt = &ktile[cur][0];
    const __bf16* vl = &vtile[cur][0];

    // ---- QK^T: two 16-key column tiles; contiguous 32B LDS B-frags ----
    v8f lg0 = {}, lg1 = {};
#pragma unroll
    for (int c = 0; c < 4; ++c) {
      v16bf b0 = *(const v16bf*)&kt[(size_t)col * 128 + c * 32 + (half << 4)];
      v16bf b1 = *(const v16bf*)&kt[(size_t)(16 + col) * 128 + c * 32 + (half << 4)];
      lg0 = wmma_bf16(qa[c], b0, lg0);
      lg1 = wmma_bf16(qa[c], b1, lg1);
    }

    // ---- soft-cap, mask, online softmax ----
    float p0[8], p1[8];
#pragma unroll
    for (int r = 0; r < 8; ++r) {
      const int t = t0 + r + rowofs;
      float v0 = tanhf(lg0[r] * INV_SOFT_CAP_) * SOFT_CAP_;
      float v1 = tanhf(lg1[r] * INV_SOFT_CAP_) * SOFT_CAP_;
      const int sa = s0 + col, sb = s0 + 16 + col;
      v0 = ((sa <= t) && (sa > t - WINDOW_)) ? v0 : K_MASK_;
      v1 = ((sb <= t) && (sb > t - WINDOW_)) ? v1 : K_MASK_;
      float mx = fmaxf(v0, v1);
#pragma unroll
      for (int d = 1; d < 16; d <<= 1) mx = fmaxf(mx, __shfl_xor(mx, d, 32));
      const float mn = fmaxf(m[r], mx);
      const float alpha = __expf(m[r] - mn);
      p0[r] = __expf(v0 - mn);
      p1[r] = __expf(v1 - mn);
      float rs = p0[r] + p1[r];
#pragma unroll
      for (int d = 1; d < 16; d <<= 1) rs += __shfl_xor(rs, d, 32);
      l[r] = l[r] * alpha + rs;
      m[r] = mn;
#pragma unroll
      for (int j = 0; j < 8; ++j) oacc[j][r] *= alpha;
    }

    // ---- transpose P (C-layout -> A-layout) through LDS ----
#pragma unroll
    for (int r = 0; r < 8; ++r) {
      pls[(r + rowofs) * 32 + col]      = f2bf(p0[r]);
      pls[(r + rowofs) * 32 + 16 + col] = f2bf(p1[r]);
    }
    __syncthreads();
    v16bf pa = load_a_vec(&pls[(size_t)(lane & 15) * 32], akb);
    __syncthreads();

    // ---- P x V: contiguous 32B LDS B-frags from transposed V tile ----
#pragma unroll
    for (int j = 0; j < 8; ++j) {
      v16bf bv = *(const v16bf*)&vl[(size_t)(j * 16 + col) * 32 + (half << 4)];
      oacc[j] = wmma_bf16(pa, bv, oacc[j]);
    }
  }

  // ---- normalize + store encoded (bf16) ----
#pragma unroll
  for (int r = 0; r < 8; ++r) {
    const float inv = 1.0f / l[r];
    const int t = t0 + r + rowofs;
#pragma unroll
    for (int j = 0; j < 8; ++j)
      eb[((size_t)t * N_ + n) * H_ + j * 16 + col] = f2bf(oacc[j][r] * inv);
  }
}

// ---------------- kernel 3: output projection ----------------
// WoT: pre-transposed bf16 [D][NH] (row stride D_ in elements = 2048).
// grid.x = T/16 row tiles, grid.y = D/128 col tiles
__global__ __launch_bounds__(32) void outproj_kernel(
    const __bf16* __restrict__ eb, const __bf16* __restrict__ wot,
    float* __restrict__ out) {
  __shared__ __attribute__((aligned(32))) __bf16 wl[2][128 * 32];  // 16 KiB

  const int lane = threadIdx.x;
  const int t0 = blockIdx.x * 16;
  const int d0 = blockIdx.y * 128;
  const int col = lane & 15;
  const int khalf = (lane >> 4) << 4;
  const int akb = (lane >> 4) << 4;
  const int rowofs = (lane >> 4) << 3;
  const __bf16* arow = eb + (size_t)(t0 + (lane & 15)) * D_;
  const __bf16* Wt = wot + (size_t)d0 * D_;   // rows d0..d0+127, cols = nk

  tdm_load_2d(&wl[0][0], Wt, 1, 32, 128, D_, 128, D_, lane);

  v8f acc[8] = {};
  for (int kc = 0; kc < D_; kc += 32) {  // K dim = N*H = 2048
    const int cur = (kc >> 5) & 1;
    if (kc + 32 < D_) {
      tdm_load_2d(&wl[cur ^ 1][0], Wt + kc + 32, 1, 32, 128,
                  (unsigned)(D_ - (kc + 32)), 128, D_, lane);
      WAIT_TENSORCNT(1);
    } else {
      WAIT_TENSORCNT(0);
    }
    v16bf a = load_a_vec(arow + kc, akb);
    const __bf16* wt = &wl[cur][0];
#pragma unroll
    for (int j = 0; j < 8; ++j) {
      v16bf b = *(const v16bf*)&wt[(size_t)(j * 16 + col) * 32 + khalf];
      acc[j] = wmma_bf16(a, b, acc[j]);
    }
  }
#pragma unroll
  for (int r = 0; r < 8; ++r)
#pragma unroll
    for (int j = 0; j < 8; ++j)
      out[(size_t)(t0 + r + rowofs) * D_ + d0 + j * 16 + col] = acc[j][r];
}

// ---------------- launch ----------------
extern "C" void kernel_launch(void* const* d_in, const int* in_sizes, int n_in,
                              void* d_out, int out_size, void* d_ws, size_t ws_size,
                              hipStream_t stream) {
  (void)in_sizes; (void)n_in; (void)out_size; (void)ws_size;
  const float* x    = (const float*)d_in[0];
  // d_in[1] = segment_pos (arange), d_in[2] = attn_mask (causal) -> analytic
  const float* qk   = (const float*)d_in[3];
  const float* kvk  = (const float*)d_in[4];
  const float* wo   = (const float*)d_in[5];
  float* out        = (float*)d_out;

  char* ws = (char*)d_ws;
  __bf16* xb  = (__bf16*)(ws);                        // 8 MiB  [T][D]
  __bf16* qb  = (__bf16*)(ws + ((size_t)8  << 20));   // 8 MiB  [T][N][H]
  __bf16* kb  = (__bf16*)(ws + ((size_t)16 << 20));   // 4 MiB  [T][KH][H]
  __bf16* vt  = (__bf16*)(ws + ((size_t)20 << 20));   // 4 MiB  [KH][H][T]
  __bf16* eb  = (__bf16*)(ws + ((size_t)24 << 20));   // 8 MiB  [T][N*H]
  __bf16* wqt = (__bf16*)(ws + ((size_t)32 << 20));   // 8 MiB  [N][H][D]
  __bf16* wkt = (__bf16*)(ws + ((size_t)40 << 20));   // 4 MiB  [KH][H][D]
  __bf16* wvt = (__bf16*)(ws + ((size_t)44 << 20));   // 4 MiB  [KH][H][D]
  __bf16* wot = (__bf16*)(ws + ((size_t)48 << 20));   // 8 MiB  [D][N*H]

  cvt_bf16_kernel<<<(T_ * D_ + 255) / 256, 256, 0, stream>>>(x, xb, T_ * D_);
  // weight transpose+convert (one-time): src [b][R][C] f32 -> dst [b][C][R] bf16
  cvtT_bf16_kernel<<<(N_ * D_ * H_ + 255) / 256, 256, 0, stream>>>(qk, wqt, D_, H_, N_);
  cvtT_bf16_kernel<<<(KH_ * D_ * H_ + 255) / 256, 256, 0, stream>>>(kvk, wkt, D_, H_, KH_);
  cvtT_bf16_kernel<<<(KH_ * D_ * H_ + 255) / 256, 256, 0, stream>>>(
      kvk + (size_t)KH_ * D_ * H_, wvt, D_, H_, KH_);
  cvtT_bf16_kernel<<<(D_ * D_ + 255) / 256, 256, 0, stream>>>(wo, wot, D_, D_, 1);

  proj_kernel<<<dim3(T_ / 16, N_ + 2 * KH_), 32, 0, stream>>>(xb, wqt, wkt, wvt, qb, kb, vt);
  attn_kernel<<<dim3(T_ / 16, N_), 32, 0, stream>>>(qb, kb, vt, eb);
  outproj_kernel<<<dim3(T_ / 16, D_ / 128), 32, 0, stream>>>(eb, wot, out);
}
